// GCN_11484742549906
// MI455X (gfx1250) — matrix operand order repaired and checked by reading, not compiled, not run on hardware
//
#include <hip/hip_runtime.h>
#include <hip/hip_bf16.h>

// ---------------------------------------------------------------------------
// 3-layer GCN funnel for MI455X (gfx1250, wave32).
//
// Roofline: aggregation gathers dominate (~2.8GB @ 23.3TB/s ~ 120us); GEMMs
// are 33 GFLOP (only rows < n_dst matter since dis[i>=n_dst]==0 exactly in the
// reference) -> split-bf16 v_wmma_f32_16x16x32_bf16 (3 WMMAs per k-tile:
// hi*hi + hi*lo + lo*hi, fp32 accumulate => ~1e-4 rel err).
// f32->bf16 hi/lo splitting is done in separate bandwidth-bound pre-passes
// (weights into a transposed lane-contiguous layout, inputs row-major with
// ReLU folded in) so the GEMM inner loop is pure b128 loads + WMMA instead of
// being VALU-bound on conversions.
// Aggregation uses CSR-by-target (1 atomic/edge binning, atomic-free hot loop,
// coalesced 128B row gathers) instead of ~700M scattered float atomics.
// ---------------------------------------------------------------------------

typedef __attribute__((ext_vector_type(8)))  float          v8f;
typedef __attribute__((ext_vector_type(16))) __bf16         v16bf;
typedef __attribute__((ext_vector_type(16))) unsigned short v16us;
typedef __attribute__((ext_vector_type(8)))  unsigned short v8us;

#define F_HID 256   // hidden/out feature width (all layer outputs are 256)

__device__ __forceinline__ unsigned short f2bf_rne(float f) {
  unsigned u = __builtin_bit_cast(unsigned, f);
  unsigned r = u + 0x7FFFu + ((u >> 16) & 1u);   // round-to-nearest-even
  return (unsigned short)(r >> 16);
}
__device__ __forceinline__ float bf2f(unsigned short b) {
  unsigned u = ((unsigned)b) << 16;
  return __builtin_bit_cast(float, u);
}
__device__ __forceinline__ v16bf us2bf(v16us u) {
  union { v16us u; v16bf b; } c; c.u = u; return c.b;
}

// ---------------------------------------------------------------------------
// utility kernels
// ---------------------------------------------------------------------------
__global__ void __launch_bounds__(256) zero_u32(unsigned* p, int n) {
  int i = blockIdx.x * blockDim.x + threadIdx.x;
  if (i < n) p[i] = 0u;
}

// count ALL edges per target (for degree norm) and FILTERED edges (row<n_dst,
// the only ones with nonzero message) per target for CSR sizing.
__global__ void __launch_bounds__(256)
count_edges(const int* __restrict__ row, const int* __restrict__ col, int E,
            int n_dst, unsigned* __restrict__ deg_all, unsigned* __restrict__ cnt) {
  for (int e = blockIdx.x * blockDim.x + threadIdx.x; e < E;
       e += gridDim.x * blockDim.x) {
    int c = col[e];
    atomicAdd(&deg_all[c], 1u);
    if (row[e] < n_dst) atomicAdd(&cnt[c], 1u);
  }
}

// single-workgroup chunked exclusive scan (n <= ~200k, ~200 LDS passes)
__global__ void __launch_bounds__(1024)
scan_excl(const unsigned* __restrict__ cnt, unsigned* __restrict__ rowptr, int n) {
  __shared__ unsigned sdata[1024];
  __shared__ unsigned carry;
  if (threadIdx.x == 0) carry = 0u;
  __syncthreads();
  for (int base = 0; base < n; base += 1024) {
    int i = base + (int)threadIdx.x;
    unsigned v = (i < n) ? cnt[i] : 0u;
    sdata[threadIdx.x] = v;
    __syncthreads();
    #pragma unroll
    for (int off = 1; off < 1024; off <<= 1) {
      unsigned t = (threadIdx.x >= (unsigned)off) ? sdata[threadIdx.x - off] : 0u;
      __syncthreads();
      sdata[threadIdx.x] += t;
      __syncthreads();
    }
    if (i < n) rowptr[i] = carry + sdata[threadIdx.x] - v;  // exclusive
    __syncthreads();
    if (threadIdx.x == 1023) carry += sdata[1023];
    __syncthreads();
  }
  if (threadIdx.x == 0) rowptr[n] = carry;
}

// dis[i] = rsqrt(deg_all[i] + fill) for i < n_dst (always > 0); cursor = rowptr
__global__ void __launch_bounds__(256)
dis_cursor(const unsigned* __restrict__ deg_all, const unsigned* __restrict__ rowptr,
           float* __restrict__ dis, unsigned* __restrict__ cursor, int n, float fill) {
  int i = blockIdx.x * blockDim.x + threadIdx.x;
  if (i < n) {
    dis[i]    = rsqrtf((float)deg_all[i] + fill);
    cursor[i] = rowptr[i];
  }
}

// bin contributing edges (row < n_dst) into CSR: 1 atomic per edge
__global__ void __launch_bounds__(256)
scatter_edges(const int* __restrict__ row, const int* __restrict__ col, int E,
              int n_dst, unsigned* __restrict__ cursor, int* __restrict__ csr) {
  for (int e = blockIdx.x * blockDim.x + threadIdx.x; e < E;
       e += gridDim.x * blockDim.x) {
    int r = row[e];
    if (r < n_dst) {
      unsigned p = atomicAdd(&cursor[col[e]], 1u);
      csr[p] = r;
    }
  }
}

// ---------------------------------------------------------------------------
// pre-pass: split W[f_in, f_out] (row-major f32) into TRANSPOSED bf16 hi/lo
// arrays Wt[n*f_in + k] so a lane's 16 consecutive K-values are one aligned
// 32B load in the GEMM.
// ---------------------------------------------------------------------------
__global__ void __launch_bounds__(256)
split_weights(const float* __restrict__ W, unsigned short* __restrict__ whi,
              unsigned short* __restrict__ wlo, int f_in, int f_out) {
  int idx = blockIdx.x * blockDim.x + threadIdx.x;   // over f_out * f_in
  if (idx < f_in * f_out) {
    int n = idx / f_in, k = idx - n * f_in;
    float v = W[(size_t)k * f_out + n];
    unsigned short h = f2bf_rne(v);
    whi[idx] = h;
    wlo[idx] = f2bf_rne(v - bf2f(h));
  }
}

// pre-pass: split layer input (optionally ReLU'd) into row-major bf16 hi/lo
__global__ void __launch_bounds__(256)
split_relu(const float* __restrict__ X, unsigned short* __restrict__ xhi,
           unsigned short* __restrict__ xlo, long long n, int relu) {
  long long i = (long long)blockIdx.x * blockDim.x + threadIdx.x;
  if (i < n) {
    float v = X[i];
    if (relu) v = fmaxf(v, 0.0f);
    unsigned short h = f2bf_rne(v);
    xhi[i] = h;
    xlo[i] = f2bf_rne(v - bf2f(h));
  }
}

// ---------------------------------------------------------------------------
// WMMA GEMM: H[0:n_rows, 0:256] = Xsplit @ Wsplit
// block = 128 threads = 4 waves; block owns a 16-row tile; wave w owns
// 64 output columns (4 16x16 tiles). 3 WMMAs per (k-tile, n-tile).
// Inner loop: aligned b128 loads of pre-split bf16 + v_wmma only.
// ---------------------------------------------------------------------------
__global__ void __launch_bounds__(128)
gemm_wmma_splitbf16(const unsigned short* __restrict__ Xhi,
                    const unsigned short* __restrict__ Xlo,
                    const unsigned short* __restrict__ Whi,   // transposed [f_out][f_in]
                    const unsigned short* __restrict__ Wlo,
                    float* __restrict__ H, int n_rows, int f_in, int f_out) {
  const int lane = threadIdx.x & 31;
  const int wave = threadIdx.x >> 5;        // 0..3
  const int m0   = blockIdx.x * 16;
  const int n0   = wave * 64;               // 4 N-tiles of 16
  const int hi   = lane >> 4;               // lane-group select (ISA layout)

  int arow = m0 + (lane & 15);
  if (arow >= n_rows) arow = n_rows - 1;    // safe clamp (n_rows % 16 == 0 here)
  const unsigned short* __restrict__ axh = Xhi + (size_t)arow * f_in;
  const unsigned short* __restrict__ axl = Xlo + (size_t)arow * f_in;

  v8f acc[4];
  #pragma unroll
  for (int t = 0; t < 4; ++t) acc[t] = (v8f){};

  for (int k0 = 0; k0 < f_in; k0 += 32) {
    // ---- A fragment: K(i) = hi*16 + i: two aligned 16B runs
    const int r1 = k0 + hi * 16;
    const int r2 = k0 + hi * 16 + 8;
    v8us ah0 = *(const v8us*)(axh + r1);
    v8us ah1 = *(const v8us*)(axh + r2);
    v8us al0 = *(const v8us*)(axl + r1);
    v8us al1 = *(const v8us*)(axl + r2);
    const v16bf a_hi = us2bf(__builtin_shufflevector(
        ah0, ah1, 0, 1, 2, 3, 4, 5, 6, 7, 8, 9, 10, 11, 12, 13, 14, 15));
    const v16bf a_lo = us2bf(__builtin_shufflevector(
        al0, al1, 0, 1, 2, 3, 4, 5, 6, 7, 8, 9, 10, 11, 12, 13, 14, 15));

    // ---- 4 N-tiles: B fragment K(i) = hi*16 + i at column n: one 32B load
    #pragma unroll
    for (int t = 0; t < 4; ++t) {
      const size_t boff = (size_t)(n0 + t * 16 + (lane & 15)) * f_in + k0 + hi * 16;
      const v16bf b_hi = us2bf(*(const v16us*)(Whi + boff));
      const v16bf b_lo = us2bf(*(const v16us*)(Wlo + boff));

      acc[t] = __builtin_amdgcn_wmma_f32_16x16x32_bf16(
          false, a_hi, false, b_hi, (short)0, acc[t], false, false);
      acc[t] = __builtin_amdgcn_wmma_f32_16x16x32_bf16(
          false, a_hi, false, b_lo, (short)0, acc[t], false, false);
      acc[t] = __builtin_amdgcn_wmma_f32_16x16x32_bf16(
          false, a_lo, false, b_hi, (short)0, acc[t], false, false);
    }
  }

  // ---- store C/D: VGPR r -> M = r + hi*8, N = lane&15 ----
  #pragma unroll
  for (int r = 0; r < 8; ++r) {
    int row = m0 + r + hi * 8;
    if (row < n_rows) {
      float* o = H + (size_t)row * f_out + n0 + (lane & 15);
      o[0]  = acc[0][r];
      o[16] = acc[1][r];
      o[32] = acc[2][r];
      o[48] = acc[3][r];
    }
  }
}

// ---------------------------------------------------------------------------
// Atomic-free aggregation: one wave per target node, F=256 (8 f32/lane).
// out[c] = b + fill*dis[c]^2*h[c] + sum_{r in csr[c]} dis[r]*dis[c]*h[r]
// ---------------------------------------------------------------------------
__global__ void __launch_bounds__(256)
aggregate(const float* __restrict__ h, const int* __restrict__ csr,
          const unsigned* __restrict__ rowptr, const float* __restrict__ dis,
          const float* __restrict__ bias, float* __restrict__ out,
          int n_dst, float fill) {
  const int lane = threadIdx.x & 31;
  const int c = blockIdx.x * (blockDim.x >> 5) + (threadIdx.x >> 5);
  if (c >= n_dst) return;

  const float dc = dis[c];
  const float sw = fill * dc * dc;
  const float* __restrict__ hc = h + (size_t)c * F_HID;
  float acc[8];
  #pragma unroll
  for (int j = 0; j < 8; ++j)
    acc[j] = bias[lane + 32 * j] + sw * hc[lane + 32 * j];

  const unsigned p0 = rowptr[c], p1 = rowptr[c + 1];
  for (unsigned p = p0; p < p1; ++p) {
    const int r = csr[p];
    const float w = dis[r] * dc;
    const float* __restrict__ hr = h + (size_t)r * F_HID;
    #pragma unroll
    for (int j = 0; j < 8; ++j) acc[j] += w * hr[lane + 32 * j];
  }
  float* __restrict__ oc = out + (size_t)c * F_HID;
  #pragma unroll
  for (int j = 0; j < 8; ++j) oc[lane + 32 * j] = acc[j];
}

// ---------------------------------------------------------------------------
// host side
// ---------------------------------------------------------------------------
static void run_layer(const float* x_in, int relu_in, int f_in,
                      const int* erow, const int* ecol, int E,
                      const float* W, const float* bias, int n_dst, float fill,
                      float* h_buf, float* out_buf,
                      unsigned short* xhi, unsigned short* xlo,
                      unsigned short* whi, unsigned short* wlo,
                      unsigned* deg_all, unsigned* cnt, unsigned* rowptr,
                      unsigned* cursor, float* dis, int* csr,
                      hipStream_t stream) {
  dim3 b256(256);
  zero_u32<<<dim3((n_dst + 255) / 256), b256, 0, stream>>>(deg_all, n_dst);
  zero_u32<<<dim3((n_dst + 255) / 256), b256, 0, stream>>>(cnt, n_dst);
  count_edges<<<dim3((E + 255) / 256), b256, 0, stream>>>(erow, ecol, E, n_dst,
                                                          deg_all, cnt);
  scan_excl<<<dim3(1), dim3(1024), 0, stream>>>(cnt, rowptr, n_dst);
  dis_cursor<<<dim3((n_dst + 255) / 256), b256, 0, stream>>>(deg_all, rowptr,
                                                             dis, cursor, n_dst, fill);
  scatter_edges<<<dim3((E + 255) / 256), b256, 0, stream>>>(erow, ecol, E, n_dst,
                                                            cursor, csr);
  // bf16 hi/lo pre-splits (bandwidth-bound, removes all VALU from GEMM loop)
  split_weights<<<dim3((f_in * F_HID + 255) / 256), b256, 0, stream>>>(
      W, whi, wlo, f_in, F_HID);
  long long nx = (long long)n_dst * f_in;
  split_relu<<<dim3((unsigned)((nx + 255) / 256)), b256, 0, stream>>>(
      x_in, xhi, xlo, nx, relu_in);
  gemm_wmma_splitbf16<<<dim3((n_dst + 15) / 16), dim3(128), 0, stream>>>(
      xhi, xlo, whi, wlo, h_buf, n_dst, f_in, F_HID);
  aggregate<<<dim3((n_dst + 7) / 8), b256, 0, stream>>>(h_buf, csr, rowptr, dis,
                                                        bias, out_buf, n_dst, fill);
}

extern "C" void kernel_launch(void* const* d_in, const int* in_sizes, int n_in,
                              void* d_out, int out_size, void* d_ws, size_t ws_size,
                              hipStream_t stream) {
  const float* x   = (const float*)d_in[0];
  const int*   ei0 = (const int*)d_in[1];
  const int*   ei1 = (const int*)d_in[2];
  const int*   ei2 = (const int*)d_in[3];
  const float* W0  = (const float*)d_in[4];
  const float* b0  = (const float*)d_in[5];
  const float* W1  = (const float*)d_in[6];
  const float* b1  = (const float*)d_in[7];
  const float* W2  = (const float*)d_in[8];
  const float* b2  = (const float*)d_in[9];

  const int E0 = in_sizes[1] / 2;           // edge_index is [2, E]
  const int E1 = in_sizes[2] / 2;
  const int E2 = in_sizes[3] / 2;
  const int DEG = 16;
  const int n1 = E0 / DEG;                  // 200000
  const int n2 = E1 / DEG;                  // 100000
  const int n3 = out_size / F_HID;          // 50000
  const int f_in0 = in_sizes[4] / F_HID;    // 128 (W0 is [F_IN, 256])

  // workspace carve-up (max n_dst = n1, max E = E0)
  size_t off = 0;
  auto carve = [&](size_t bytes) -> void* {
    void* p = (char*)d_ws + off;
    off += (bytes + 255) & ~(size_t)255;
    return p;
  };
  float*    bufA    = (float*)carve((size_t)n1 * F_HID * sizeof(float));  // h
  float*    bufB    = (float*)carve((size_t)n1 * F_HID * sizeof(float));  // y
  unsigned short* xhi = (unsigned short*)carve((size_t)n1 * F_HID * 2);
  unsigned short* xlo = (unsigned short*)carve((size_t)n1 * F_HID * 2);
  unsigned short* whi = (unsigned short*)carve((size_t)F_HID * F_HID * 2);
  unsigned short* wlo = (unsigned short*)carve((size_t)F_HID * F_HID * 2);
  unsigned* deg_all = (unsigned*)carve((size_t)n1 * 4);
  unsigned* cnt     = (unsigned*)carve((size_t)n1 * 4);
  unsigned* rowptr  = (unsigned*)carve((size_t)(n1 + 1) * 4);
  unsigned* cursor  = (unsigned*)carve((size_t)n1 * 4);
  float*    dis     = (float*)carve((size_t)n1 * 4);
  int*      csr     = (int*)carve((size_t)E0 * 4);
  (void)ws_size; (void)n_in;

  // layer 0: x -> h(bufA) -> y0(bufB), fill=2, no input relu
  run_layer(x,    0, f_in0, ei0, ei0 + E0, E0, W0, b0, n1, 2.0f,
            bufA, bufB, xhi, xlo, whi, wlo,
            deg_all, cnt, rowptr, cursor, dis, csr, stream);
  // layer 1: relu(y0)(bufB) -> h(bufA) -> y1(bufB prefix, y0 dead), fill=2
  run_layer(bufB, 1, F_HID, ei1, ei1 + E1, E1, W1, b1, n2, 2.0f,
            bufA, bufB, xhi, xlo, whi, wlo,
            deg_all, cnt, rowptr, cursor, dis, csr, stream);
  // layer 2: relu(y1)(bufB) -> h(bufA) -> d_out, fill=1 (improved=False)
  run_layer(bufB, 1, F_HID, ei2, ei2 + E2, E2, W2, b2, n3, 1.0f,
            bufA, (float*)d_out, xhi, xlo, whi, wlo,
            deg_all, cnt, rowptr, cursor, dis, csr, stream);
}
